// LatentAttentionBlock_32366873542900
// MI455X (gfx1250) — compile-verified, matrix operand order checked by reading, not run on hardware
//
#include <hip/hip_runtime.h>
#include <math.h>

// ---------------------------------------------------------------------------
// Types
// ---------------------------------------------------------------------------
typedef __attribute__((ext_vector_type(16))) __bf16 bf16x16;
typedef __attribute__((ext_vector_type(8)))  __bf16 bf16x8;
typedef __attribute__((ext_vector_type(8)))  float  f32x8;

// v4i matching the async-LDS builtin signature (vector_size(16) int)
typedef int v4i __attribute__((vector_size(16)));
typedef __attribute__((address_space(1))) v4i gas_v4i;   // global
typedef __attribute__((address_space(3))) v4i las_v4i;   // LDS

// Async global->LDS path (gfx1250 GLOBAL_LOAD_ASYNC_TO_LDS_*, ASYNCcnt)
#if defined(__has_builtin)
#  if __has_builtin(__builtin_amdgcn_global_load_async_to_lds_b128) && \
      __has_builtin(__builtin_amdgcn_s_wait_asynccnt)
#    define HAVE_ASYNC_LDS 1
#  endif
#endif
#ifndef HAVE_ASYNC_LDS
#  define HAVE_ASYNC_LDS 0
#endif

// pointer -> integer -> AS-qualified pointer (always-valid casts; on amdgcn
// the generic LDS address carries the LDS offset in its low 32 bits)
__device__ __forceinline__ gas_v4i* as_gv4(const void* p) {
    return (gas_v4i*)(unsigned long long)(size_t)p;
}
__device__ __forceinline__ las_v4i* as_lv4(void* p) {
    return (las_v4i*)(unsigned int)(size_t)p;
}

// ---------------------------------------------------------------------------
// Problem constants
// ---------------------------------------------------------------------------
#define DIMN   1024
#define SEQ    1024
#define BATCH  16
#define QKD    128
#define EXPD   2048
#define NEXP   4352                 // 2*QK + 2*EXP columns of expand
#define MROWS  (BATCH * SEQ)        // 16384 token rows

// workspace layout (bytes)
#define H_OFF      ((size_t)0)                            // h bf16        33554432
#define WEXP_OFF   ((size_t)33554432)                     // Wexp bf16      8912896
#define WPROJ_OFF  ((size_t)42467328)                     // Wproj bf16     4194304
#define PROJ_OFF   ((size_t)46661632)                     // proj bf16    142606336
#define GG_OFF     ((size_t)189267968)                    // geglu bf16    67108864
#define ATTN_OFF   ((size_t)256376832)                    // attn out bf16 33554432

// ---------------------------------------------------------------------------
// Helpers
// ---------------------------------------------------------------------------
__device__ __forceinline__ bf16x16 ldfrag(const __bf16* p0, const __bf16* p1) {
    bf16x8 lo = *(const bf16x8*)p0;
    bf16x8 hi = *(const bf16x8*)p1;
    bf16x16 r;
#pragma unroll
    for (int i = 0; i < 8; ++i) { r[i] = lo[i]; r[i + 8] = hi[i]; }
    return r;
}

__device__ __forceinline__ f32x8 wmma_bf16(bf16x16 a, bf16x16 b, f32x8 c) {
    return __builtin_amdgcn_wmma_f32_16x16x32_bf16(
        /*neg_a=*/false, a, /*neg_b=*/false, b,
        /*c_mod=*/(short)0, c, /*reuse_a=*/false, /*reuse_b=*/false);
}

// ---------------------------------------------------------------------------
// 1) LayerNorm -> bf16 activations (one row of 1024 per block of 256)
// ---------------------------------------------------------------------------
__global__ void ln_kernel(const float* __restrict__ x,
                          const float* __restrict__ nw,
                          __bf16* __restrict__ h) {
    const int row = blockIdx.x;
    const int tid = threadIdx.x;
    const float4 v = ((const float4*)(x + (size_t)row * DIMN))[tid];

    float s  = v.x + v.y + v.z + v.w;
    float sq = v.x * v.x + v.y * v.y + v.z * v.z + v.w * v.w;
#pragma unroll
    for (int m = 16; m >= 1; m >>= 1) {
        s  += __shfl_xor(s, m, 32);
        sq += __shfl_xor(sq, m, 32);
    }
    __shared__ float ss[8], sqs[8], stats[2];
    const int wid = tid >> 5, lane = tid & 31;
    if (lane == 0) { ss[wid] = s; sqs[wid] = sq; }
    __syncthreads();
    if (tid == 0) {
        float ts = 0.f, tq = 0.f;
#pragma unroll
        for (int i = 0; i < 8; ++i) { ts += ss[i]; tq += sqs[i]; }
        const float mean = ts * (1.0f / DIMN);
        const float var  = tq * (1.0f / DIMN) - mean * mean;
        stats[0] = mean;
        stats[1] = rsqrtf(var + 1e-5f);
    }
    __syncthreads();
    const float mean = stats[0], rstd = stats[1];
    const float4 w4 = ((const float4*)nw)[tid];
    __bf16* hr = h + (size_t)row * DIMN + tid * 4;
    hr[0] = (__bf16)((v.x - mean) * rstd * w4.x);
    hr[1] = (__bf16)((v.y - mean) * rstd * w4.y);
    hr[2] = (__bf16)((v.z - mean) * rstd * w4.z);
    hr[3] = (__bf16)((v.w - mean) * rstd * w4.w);
}

// ---------------------------------------------------------------------------
// 2) Weight downcast fp32 -> bf16
// ---------------------------------------------------------------------------
#define NWEXP (NEXP * DIMN)    // 4456448
#define NWPRJ (DIMN * EXPD)    // 2097152
__global__ void cast_w_kernel(const float* __restrict__ we,
                              const float* __restrict__ wp,
                              __bf16* __restrict__ oe,
                              __bf16* __restrict__ op) {
    const size_t i = (size_t)blockIdx.x * blockDim.x + threadIdx.x;
    if (i < (size_t)NWEXP) {
        oe[i] = (__bf16)we[i];
    } else if (i < (size_t)NWEXP + NWPRJ) {
        op[i - NWEXP] = (__bf16)wp[i - NWEXP];
    }
}

// ---------------------------------------------------------------------------
// 3) Expand GEMM: proj = h @ Wexp^T   (M=16384, N=4352, K=1024), bf16 WMMA
//    block = 256 threads (8 waves), 128x128 tile, K-step 32,
//    async global->LDS staging when available
// ---------------------------------------------------------------------------
__global__ void gemm_expand(const __bf16* __restrict__ A,    // h, lda=1024
                            const __bf16* __restrict__ B,    // Wexp, ldb=1024
                            __bf16* __restrict__ C) {        // proj, ldc=4352
    __shared__ __bf16 As[128][40];
    __shared__ __bf16 Bs[128][40];

    const int m0 = blockIdx.x * 128;
    const int n0 = blockIdx.y * 128;
    const int tid = threadIdx.x;
    const int w = tid >> 5, l = tid & 31;
    const int lh = l >> 4, ln_ = l & 15;
    const int rA = (w & 3) * 32;     // wave row group
    const int cB = (w >> 2) * 64;    // wave col group

    f32x8 acc[2][4] = {};

    const int trow = tid >> 1, thalf = tid & 1;
    for (int k0 = 0; k0 < DIMN; k0 += 32) {
        const __bf16* ga = A + (size_t)(m0 + trow) * DIMN + k0 + thalf * 16;
        const __bf16* gb = B + (size_t)(n0 + trow) * DIMN + k0 + thalf * 16;
        if (k0 + 32 < DIMN) {
            __builtin_prefetch(ga + 32, 0, 3);
            __builtin_prefetch(gb + 32, 0, 3);
        }
#if HAVE_ASYNC_LDS
        __syncthreads();                         // prior fragment reads done
        __builtin_amdgcn_global_load_async_to_lds_b128(
            as_gv4(ga), as_lv4(&As[trow][thalf * 16]), 0, 0);
        __builtin_amdgcn_global_load_async_to_lds_b128(
            as_gv4(ga), as_lv4(&As[trow][thalf * 16]), 16, 0);
        __builtin_amdgcn_global_load_async_to_lds_b128(
            as_gv4(gb), as_lv4(&Bs[trow][thalf * 16]), 0, 0);
        __builtin_amdgcn_global_load_async_to_lds_b128(
            as_gv4(gb), as_lv4(&Bs[trow][thalf * 16]), 16, 0);
        __builtin_amdgcn_s_wait_asynccnt(0);
        __syncthreads();
#else
        bf16x8 a0 = *(const bf16x8*)ga, a1 = *(const bf16x8*)(ga + 8);
        bf16x8 b0 = *(const bf16x8*)gb, b1 = *(const bf16x8*)(gb + 8);
        __syncthreads();
        *(bf16x8*)&As[trow][thalf * 16]     = a0;
        *(bf16x8*)&As[trow][thalf * 16 + 8] = a1;
        *(bf16x8*)&Bs[trow][thalf * 16]     = b0;
        *(bf16x8*)&Bs[trow][thalf * 16 + 8] = b1;
        __syncthreads();
#endif

        bf16x16 af[2], bfg[4];
#pragma unroll
        for (int i = 0; i < 2; ++i) {
            const int r = rA + i * 16 + ln_;
            af[i] = ldfrag(&As[r][lh * 8], &As[r][16 + lh * 8]);
        }
#pragma unroll
        for (int j = 0; j < 4; ++j) {
            const int r = cB + j * 16 + ln_;
            bfg[j] = ldfrag(&Bs[r][lh * 8], &Bs[r][16 + lh * 8]);
        }
#pragma unroll
        for (int i = 0; i < 2; ++i)
#pragma unroll
            for (int j = 0; j < 4; ++j)
                acc[i][j] = wmma_bf16(af[i], bfg[j], acc[i][j]);
    }

#pragma unroll
    for (int i = 0; i < 2; ++i)
#pragma unroll
        for (int j = 0; j < 4; ++j)
#pragma unroll
            for (int r = 0; r < 8; ++r) {
                const int mr = m0 + rA + i * 16 + r + 8 * lh;
                const int nc = n0 + cB + j * 16 + ln_;
                C[(size_t)mr * NEXP + nc] = (__bf16)acc[i][j][r];
            }
}

// ---------------------------------------------------------------------------
// 4) GEGLU: gg[row, j] = lin * gelu_exact(pre)  (16384 x 2048)
// ---------------------------------------------------------------------------
__global__ void geglu_kernel(const __bf16* __restrict__ proj,
                             __bf16* __restrict__ gg) {
    const size_t i = (size_t)blockIdx.x * blockDim.x + threadIdx.x;
    const size_t row = i >> 11;
    const int    j   = (int)(i & 2047);
    const float lin = (float)proj[row * NEXP + 2 * QKD + j];
    const float pre = (float)proj[row * NEXP + 2 * QKD + EXPD + j];
    const float g   = 0.5f * pre * (1.0f + erff(pre * 0.70710678118654752f));
    gg[i] = (__bf16)(lin * g);
}

// ---------------------------------------------------------------------------
// 5) Causal flash attention (16 query rows / block; 8 waves x 128 V cols)
//    scores: q @ k^T via WMMA; online softmax; PV via WMMA with LDS-transposed V
// ---------------------------------------------------------------------------
__global__ void attn_kernel(const __bf16* __restrict__ proj,   // q@0, k@128
                            const __bf16* __restrict__ gg,     // gv = gg[:,1024:]
                            const float* __restrict__ pbm,
                            __bf16* __restrict__ attnb) {
    __shared__ __bf16 VT[1024][32];        // transposed V tile: [col][key]
    __shared__ __bf16 Pt[8][16][32];       // per-wave softmax probs

    const int qbase = blockIdx.x * 16;
    const int b     = blockIdx.y;
    const int tid   = threadIdx.x;
    const int w = tid >> 5, l = tid & 31;
    const int lh = l >> 4, ln_ = l & 15;

    const float sp    = log1pf(expf(pbm[0]));      // softplus(pos_bias_mult)
    const float invsq = 0.08838834764831845f;      // 1/sqrt(128)

    const __bf16* projB = proj + (size_t)b * SEQ * NEXP;
    const __bf16* ggB   = gg   + (size_t)b * SEQ * EXPD;

    // Q fragments (16 x 128 over K), A layout
    bf16x16 qa[4];
    {
        const __bf16* pr = projB + (size_t)(qbase + ln_) * NEXP;
#pragma unroll
        for (int s = 0; s < 4; ++s)
            qa[s] = ldfrag(pr + 32 * s + lh * 8, pr + 32 * s + 16 + lh * 8);
    }

    f32x8 accv[8] = {};
    float mrun[8], srun[8];
#pragma unroll
    for (int r = 0; r < 8; ++r) { mrun[r] = -3.0e38f; srun[r] = 0.f; }

    const int KB = (qbase + 15) / 32 + 1;          // 32-key blocks needed
    for (int kb = 0; kb < KB; ++kb) {
        const int k0 = kb * 32;
        __syncthreads();                           // prior VT reads done
        // ---- stage V (32 keys x 1024 cols), transposed into LDS ----
        {
            const int half = tid >> 7;             // 0/1
            const int col0 = (tid & 127) * 8;
#pragma unroll
            for (int it = 0; it < 16; ++it) {
                const int row = 2 * it + half;
                bf16x8 d = *(const bf16x8*)(ggB + (size_t)(k0 + row) * EXPD +
                                            (EXPD - DIMN) + col0);
#pragma unroll
                for (int jj = 0; jj < 8; ++jj) VT[col0 + jj][row] = d[jj];
            }
        }
        // ---- scores: two 16x16 tiles, K=128 ----
        f32x8 c0 = {}, c1 = {};
#pragma unroll
        for (int s = 0; s < 4; ++s) {
            const __bf16* kr0 = projB + (size_t)(k0 + ln_) * NEXP + QKD;
            const __bf16* kr1 = projB + (size_t)(k0 + 16 + ln_) * NEXP + QKD;
            bf16x16 kb0 = ldfrag(kr0 + 32 * s + lh * 8, kr0 + 32 * s + 16 + lh * 8);
            bf16x16 kb1 = ldfrag(kr1 + 32 * s + lh * 8, kr1 + 32 * s + 16 + lh * 8);
            c0 = wmma_bf16(qa[s], kb0, c0);
            c1 = wmma_bf16(qa[s], kb1, c1);
        }
        // ---- bias + online softmax ----
        float scl[8];
#pragma unroll
        for (int r = 0; r < 8; ++r) {
            const int m = r + 8 * lh;
            const int qidx = qbase + m;
            const int ki0 = k0 + ln_, ki1 = k0 + 16 + ln_;
            float v0 = c0[r] * invsq +
                       (ki0 <= qidx ? sp * (float)(ki0 - qidx) : -3.0e38f);
            float v1 = c1[r] * invsq +
                       (ki1 <= qidx ? sp * (float)(ki1 - qidx) : -3.0e38f);
            float bm = fmaxf(v0, v1);
#pragma unroll
            for (int msk = 8; msk >= 1; msk >>= 1)
                bm = fmaxf(bm, __shfl_xor(bm, msk, 32));
            const float mnew = fmaxf(mrun[r], bm);
            const float sc   = expf(mrun[r] - mnew);
            const float p0 = expf(v0 - mnew);
            const float p1 = expf(v1 - mnew);
            float rs = p0 + p1;
#pragma unroll
            for (int msk = 8; msk >= 1; msk >>= 1)
                rs += __shfl_xor(rs, msk, 32);
            srun[r] = srun[r] * sc + rs;
            mrun[r] = mnew;
            scl[r]  = sc;
            Pt[w][m][ln_]      = (__bf16)p0;
            Pt[w][m][ln_ + 16] = (__bf16)p1;
        }
#pragma unroll
        for (int t = 0; t < 8; ++t)
#pragma unroll
            for (int r = 0; r < 8; ++r) accv[t][r] *= scl[r];
        __syncthreads();                           // VT + Pt visible
        // ---- PV: P(16x32) @ V(32x128 of this wave's columns) ----
        bf16x16 pa = ldfrag(&Pt[w][ln_][lh * 8], &Pt[w][ln_][16 + lh * 8]);
#pragma unroll
        for (int t = 0; t < 8; ++t) {
            const int col = w * 128 + t * 16 + ln_;
            bf16x16 vb = ldfrag(&VT[col][lh * 8], &VT[col][16 + lh * 8]);
            accv[t] = wmma_bf16(pa, vb, accv[t]);
        }
    }
    // ---- normalize + store bf16 ----
    float inv[8];
#pragma unroll
    for (int r = 0; r < 8; ++r) inv[r] = 1.0f / srun[r];
#pragma unroll
    for (int t = 0; t < 8; ++t)
#pragma unroll
        for (int r = 0; r < 8; ++r) {
            const int m = r + 8 * lh;
            const int col = w * 128 + t * 16 + ln_;
            attnb[(size_t)(b * SEQ + qbase + m) * DIMN + col] =
                (__bf16)(accv[t][r] * inv[r]);
        }
}

// ---------------------------------------------------------------------------
// 6) Project GEMM: out = concat(gl, attn) @ Wproj^T + x
//    (M=16384, N=1024, K=2048) bf16 WMMA, fp32 output with residual,
//    async global->LDS staging when available
// ---------------------------------------------------------------------------
__global__ void gemm_project(const __bf16* __restrict__ gg,      // gl = gg[:, :1024]
                             const __bf16* __restrict__ attnb,   // (16384,1024)
                             const __bf16* __restrict__ B,       // Wproj, ldb=2048
                             const float* __restrict__ x,
                             float* __restrict__ out) {
    __shared__ __bf16 As[128][40];
    __shared__ __bf16 Bs[128][40];

    const int m0 = blockIdx.x * 128;
    const int n0 = blockIdx.y * 128;
    const int tid = threadIdx.x;
    const int w = tid >> 5, l = tid & 31;
    const int lh = l >> 4, ln_ = l & 15;
    const int rA = (w & 3) * 32;
    const int cB = (w >> 2) * 64;

    f32x8 acc[2][4] = {};

    const int trow = tid >> 1, thalf = tid & 1;
    for (int k0 = 0; k0 < EXPD; k0 += 32) {
        const int kk = k0 + thalf * 16;            // never crosses 1024 boundary
        const int mr = m0 + trow;
        const __bf16* ga = (kk < DIMN)
            ? gg    + (size_t)mr * EXPD + kk
            : attnb + (size_t)mr * DIMN + (kk - DIMN);
        const __bf16* gb = B + (size_t)(n0 + trow) * EXPD + kk;
#if HAVE_ASYNC_LDS
        __syncthreads();
        __builtin_amdgcn_global_load_async_to_lds_b128(
            as_gv4(ga), as_lv4(&As[trow][thalf * 16]), 0, 0);
        __builtin_amdgcn_global_load_async_to_lds_b128(
            as_gv4(ga), as_lv4(&As[trow][thalf * 16]), 16, 0);
        __builtin_amdgcn_global_load_async_to_lds_b128(
            as_gv4(gb), as_lv4(&Bs[trow][thalf * 16]), 0, 0);
        __builtin_amdgcn_global_load_async_to_lds_b128(
            as_gv4(gb), as_lv4(&Bs[trow][thalf * 16]), 16, 0);
        __builtin_amdgcn_s_wait_asynccnt(0);
        __syncthreads();
#else
        bf16x8 a0 = *(const bf16x8*)ga, a1 = *(const bf16x8*)(ga + 8);
        bf16x8 b0 = *(const bf16x8*)gb, b1 = *(const bf16x8*)(gb + 8);
        __syncthreads();
        *(bf16x8*)&As[trow][thalf * 16]     = a0;
        *(bf16x8*)&As[trow][thalf * 16 + 8] = a1;
        *(bf16x8*)&Bs[trow][thalf * 16]     = b0;
        *(bf16x8*)&Bs[trow][thalf * 16 + 8] = b1;
        __syncthreads();
#endif

        bf16x16 af[2], bfg[4];
#pragma unroll
        for (int i = 0; i < 2; ++i) {
            const int r = rA + i * 16 + ln_;
            af[i] = ldfrag(&As[r][lh * 8], &As[r][16 + lh * 8]);
        }
#pragma unroll
        for (int j = 0; j < 4; ++j) {
            const int r = cB + j * 16 + ln_;
            bfg[j] = ldfrag(&Bs[r][lh * 8], &Bs[r][16 + lh * 8]);
        }
#pragma unroll
        for (int i = 0; i < 2; ++i)
#pragma unroll
            for (int j = 0; j < 4; ++j)
                acc[i][j] = wmma_bf16(af[i], bfg[j], acc[i][j]);
    }

#pragma unroll
    for (int i = 0; i < 2; ++i)
#pragma unroll
        for (int j = 0; j < 4; ++j)
#pragma unroll
            for (int r = 0; r < 8; ++r) {
                const int mr = m0 + rA + i * 16 + r + 8 * lh;
                const int nc = n0 + cB + j * 16 + ln_;
                out[(size_t)mr * DIMN + nc] =
                    x[(size_t)mr * DIMN + nc] + acc[i][j][r];
            }
}

// ---------------------------------------------------------------------------
// Host launcher
// ---------------------------------------------------------------------------
extern "C" void kernel_launch(void* const* d_in, const int* in_sizes, int n_in,
                              void* d_out, int out_size, void* d_ws, size_t ws_size,
                              hipStream_t stream) {
    const float* x         = (const float*)d_in[0];
    const float* norm_w    = (const float*)d_in[1];
    const float* expand_w  = (const float*)d_in[2];
    const float* project_w = (const float*)d_in[3];
    const float* pbm       = (const float*)d_in[4];
    float* out = (float*)d_out;

    char* ws = (char*)d_ws;
    __bf16* h     = (__bf16*)(ws + H_OFF);
    __bf16* wexp  = (__bf16*)(ws + WEXP_OFF);
    __bf16* wproj = (__bf16*)(ws + WPROJ_OFF);
    __bf16* proj  = (__bf16*)(ws + PROJ_OFF);
    __bf16* gg    = (__bf16*)(ws + GG_OFF);
    __bf16* attnb = (__bf16*)(ws + ATTN_OFF);

    // 1) LayerNorm -> bf16
    ln_kernel<<<MROWS, 256, 0, stream>>>(x, norm_w, h);
    // 2) weights -> bf16
    {
        const size_t n = (size_t)NWEXP + NWPRJ;
        cast_w_kernel<<<(unsigned)((n + 255) / 256), 256, 0, stream>>>(
            expand_w, project_w, wexp, wproj);
    }
    // 3) expand GEMM (WMMA)
    gemm_expand<<<dim3(MROWS / 128, NEXP / 128), 256, 0, stream>>>(h, wexp, proj);
    // 4) GEGLU
    {
        const size_t n = (size_t)MROWS * EXPD;
        geglu_kernel<<<(unsigned)(n / 256), 256, 0, stream>>>(proj, gg);
    }
    // 5) causal attention (WMMA flash)
    attn_kernel<<<dim3(SEQ / 16, BATCH), 256, 0, stream>>>(proj, gg, pbm, attnb);
    // 6) project GEMM + residual (WMMA)
    gemm_project<<<dim3(MROWS / 128, DIMN / 128), 256, 0, stream>>>(
        gg, attnb, wproj, x, out);
}
